// FrozenBNBLinear_1486058684862
// MI455X (gfx1250) — matrix operand order repaired and checked by reading, not compiled, not run on hardware
//
#include <hip/hip_runtime.h>

typedef __attribute__((ext_vector_type(16))) _Float16 v16h;
typedef __attribute__((ext_vector_type(8)))  _Float16 v8h;
typedef __attribute__((ext_vector_type(8)))  float    v8f;
typedef __attribute__((ext_vector_type(4)))  float    v4f;
typedef __attribute__((ext_vector_type(4)))  int      v4i;

// Problem dims (fixed by the reference)
constexpr int M_TOT = 128;    // B*S = 8*16
constexpr int N_TOT = 16384;  // OUT
constexpr int K_TOT = 4096;   // IN (== BLOCKSIZE -> absmax is per weight row)

// Tiling: BM = all 128 rows of x (weights streamed from HBM exactly once),
// BN = 64 -> 256 workgroups; ~31KB LDS -> 2 workgroups resident per WGP.
constexpr int BN   = 64;
constexpr int BK   = 64;      // K step per LDS stage (2 wmma K-steps)
constexpr int LSTR = BK + 16; // 80 halfs = 160B row: 16B-aligned b128 reads, bank-spread

__global__ __launch_bounds__(256)
void bnb_linear_wmma(const float* __restrict__ x,
                     const int*   __restrict__ wq,
                     const float* __restrict__ absmax,
                     const float* __restrict__ code,
                     const float* __restrict__ bias,
                     float*       __restrict__ out)
{
    __shared__ float    lut[256];
    __shared__ _Float16 As[M_TOT * LSTR];  // x tile, [m][k], f16 (20 KB)
    __shared__ _Float16 Bs[BN    * LSTR];  // dequant weight tile, [n][k], f16 (10 KB)

    const int t      = threadIdx.x;
    const int lane   = t & 31;
    const int wave   = t >> 5;
    const int wave_m = wave >> 1;   // 0..3 -> M rows [wave_m*32, +32)
    const int wave_n = wave & 1;    // 0..1 -> N cols [wave_n*32, +32)
    const int n0     = blockIdx.x * BN;

    // Stage the 256-entry dequant codebook in LDS.
    lut[t] = code[t];

    v8f acc[2][2];
    #pragma unroll
    for (int i = 0; i < 2; ++i)
        #pragma unroll
        for (int j = 0; j < 2; ++j)
            acc[i][j] = (v8f){0.f,0.f,0.f,0.f,0.f,0.f,0.f,0.f};

    // Cooperative-load coordinates: 256 threads cover 16(row) x 64(k) per pass.
    const int lrow  = t >> 4;        // 0..15
    const int lcol4 = (t & 15) * 4;  // 0,4,...,60

    // absmax is per weight row (BLOCKSIZE == IN): hoist the 4 scales out of the K loop.
    float sc[4];
    #pragma unroll
    for (int i = 0; i < 4; ++i) sc[i] = absmax[n0 + lrow + i * 16];

    const float* xp = x  + (size_t)lrow * K_TOT + lcol4;
    const int*   qp = wq + (size_t)(n0 + lrow) * K_TOT + lcol4;

    // ---- register stage for the incoming K tile (explicit double buffer) ----
    v4f xr[8];  // 128 x-rows / 16 per pass
    v4i qr[4];  //  64 w-rows / 16 per pass
    #pragma unroll
    for (int i = 0; i < 8; ++i)
        xr[i] = *(const v4f*)(xp + (size_t)i * 16 * K_TOT);
    #pragma unroll
    for (int i = 0; i < 4; ++i)
        qr[i] = *(const v4i*)(qp + (size_t)i * 16 * K_TOT);
    __syncthreads();  // lut visible before first dequant

    const int mlane = lane & 15;
    const int ahalf = (lane < 16) ? 0 : 8;   // A frag K sub-offset per ISA layout
    const int bhalf = (lane < 16) ? 0 : 8;   // B frag K sub-offset per ISA layout

    for (int kb = 0; kb < K_TOT; kb += BK) {
        // ---- staged registers -> LDS (x: f32->f16; w: LUT dequant -> f16) ----
        #pragma unroll
        for (int i = 0; i < 8; ++i) {
            const int m = lrow + i * 16;
            _Float16* ad = &As[m * LSTR + lcol4];
            ad[0] = (_Float16)xr[i].x; ad[1] = (_Float16)xr[i].y;
            ad[2] = (_Float16)xr[i].z; ad[3] = (_Float16)xr[i].w;
        }
        #pragma unroll
        for (int i = 0; i < 4; ++i) {
            const int n = lrow + i * 16;
            const float s = sc[i];
            _Float16* bd = &Bs[n * LSTR + lcol4];
            bd[0] = (_Float16)(lut[qr[i].x] * s);
            bd[1] = (_Float16)(lut[qr[i].y] * s);
            bd[2] = (_Float16)(lut[qr[i].z] * s);
            bd[3] = (_Float16)(lut[qr[i].w] * s);
        }
        __syncthreads();

        // ---- issue next tile's global loads; wait lands in next dequant phase ----
        const int kn = kb + BK;
        if (kn < K_TOT) {
            #pragma unroll
            for (int i = 0; i < 8; ++i)
                xr[i] = *(const v4f*)(xp + (size_t)i * 16 * K_TOT + kn);
            #pragma unroll
            for (int i = 0; i < 4; ++i) {
                qr[i] = *(const v4i*)(qp + (size_t)i * 16 * K_TOT + kn);
                if (kn + BK < K_TOT)  // warm L2 two tiles ahead on the weight stream
                    __builtin_prefetch(qp + (size_t)i * 16 * K_TOT + kn + BK, 0, 1);
            }
        }

        // ---- WMMA over the two K=32 sub-steps of this BK=64 stage ----
        #pragma unroll
        for (int ks = 0; ks < BK; ks += 32) {
            v16h a[2];
            #pragma unroll
            for (int fm = 0; fm < 2; ++fm) {
                const _Float16* p =
                    &As[(wave_m * 32 + fm * 16 + mlane) * LSTR + ks + ahalf];
                v8h lo = *(const v8h*)p;          // K = base .. base+7
                v8h hi = *(const v8h*)(p + 16);   // K = base+16 .. base+23
                a[fm] = __builtin_shufflevector(lo, hi,
                        0,1,2,3,4,5,6,7,8,9,10,11,12,13,14,15);
            }
            v16h b[2];
            #pragma unroll
            for (int fn = 0; fn < 2; ++fn) {
                const _Float16* p =
                    &Bs[(wave_n * 32 + fn * 16 + mlane) * LSTR + ks + bhalf];
                v8h lo = *(const v8h*)p;          // K = base .. base+7
                v8h hi = *(const v8h*)(p + 16);   // K = base+16 .. base+23
                b[fn] = __builtin_shufflevector(lo, hi,
                        0,1,2,3,4,5,6,7,8,9,10,11,12,13,14,15);
            }
            #pragma unroll
            for (int fm = 0; fm < 2; ++fm)
                #pragma unroll
                for (int fn = 0; fn < 2; ++fn)
                    acc[fm][fn] = __builtin_amdgcn_wmma_f32_16x16x32_f16(
                        /*neg_a=*/false, a[fm], /*neg_b=*/false, b[fn],
                        /*c_mod=*/(short)0, acc[fm][fn],
                        /*reuse_a=*/false, /*reuse_b=*/false);
        }
        __syncthreads();
    }

    // ---- epilogue: bias add + store per C-matrix VGPR layout ----
    const int mbase = wave_m * 32;
    const int moff  = (lane < 16) ? 0 : 8;
    const int ncol  = lane & 15;
    #pragma unroll
    for (int fm = 0; fm < 2; ++fm) {
        #pragma unroll
        for (int fn = 0; fn < 2; ++fn) {
            const int n    = n0 + wave_n * 32 + fn * 16 + ncol;
            const float bv = bias[n];
            #pragma unroll
            for (int r = 0; r < 8; ++r) {
                const int m = mbase + fm * 16 + moff + r;
                out[(size_t)m * N_TOT + n] = acc[fm][fn][r] + bv;
            }
        }
    }
}

extern "C" void kernel_launch(void* const* d_in, const int* in_sizes, int n_in,
                              void* d_out, int out_size, void* d_ws, size_t ws_size,
                              hipStream_t stream) {
    const float* x      = (const float*)d_in[0];
    const int*   wq     = (const int*)  d_in[1];
    const float* absmax = (const float*)d_in[2];
    const float* code   = (const float*)d_in[3];
    const float* bias   = (const float*)d_in[4];
    float*       out    = (float*)d_out;

    dim3 grid(N_TOT / BN);   // 256 blocks; each covers all M and 64 output cols
    dim3 block(256);         // 8 wave32 waves, ~31KB LDS -> 2 blocks per WGP
    bnb_linear_wmma<<<grid, block, 0, stream>>>(x, wq, absmax, code, bias, out);
}
